// iMOE_30331059044664
// MI455X (gfx1250) — matrix operand order repaired and checked by reading, not compiled, not run on hardware
//
#include <hip/hip_runtime.h>

// ---------------------------------------------------------------------------
// Types for CDNA5 WMMA (wave32): v_wmma_f32_16x16x32_bf16
// ---------------------------------------------------------------------------
typedef __attribute__((ext_vector_type(16))) unsigned short v16u;
typedef __attribute__((ext_vector_type(8)))  unsigned short v8u;
typedef __attribute__((ext_vector_type(16))) __bf16         v16bf;
typedef __attribute__((ext_vector_type(8)))  float          v8f;

// Workspace layout (bf16 elements), all weights pre-swizzled to WMMA-B frag order
#define WS_PSW 0u
#define WS_PMW 65536u
#define WS_C1W 131072u
#define WS_C2W 163840u
#define WS_EW1 167936u      // 8 * 512*256
#define WS_EW2 1216512u     // 8 * 256*64

// Dynamic LDS layout (bytes)
#define L_TEMP   0u         // ushort[32][520]  temp (bf16) : 33280
#define L_STAGE  33280u     // ushort[32][264]  staging/eh/h : 16896
#define L_EO     50176u     // ushort[8][32*64] expert out bf16 : 32768
#define L_SIMW   82944u     // float[32]
#define L_AVE    83072u     // float[256]
#define L_MAX    84096u     // float[256]
#define L_WTS    85120u     // float[256]
#define SMEM_BYTES 86144u

// ---------------------------------------------------------------------------
// bf16 helpers (bit-level, no dependence on __bf16 conversion codegen)
// ---------------------------------------------------------------------------
static __device__ __forceinline__ unsigned short f2bf(float f) {
  unsigned u = __builtin_bit_cast(unsigned, f);
  u = u + 0x7FFFu + ((u >> 16) & 1u);       // round-to-nearest-even
  return (unsigned short)(u >> 16);
}
static __device__ __forceinline__ float bf2f(unsigned short s) {
  return __builtin_bit_cast(float, (unsigned)s << 16);
}

static __device__ __forceinline__ v8f wmma_bf16(v16bf a, v16bf b, v8f c) {
  return __builtin_amdgcn_wmma_f32_16x16x32_bf16(false, a, false, b, (short)0, c,
                                                 false, false);
}

// A fragment (16x32 bf16, M x K) from LDS (row-major, stride in halves).
// ISA layout: lane l holds row M=l&15; halves 0..7 = K k0..k0+7, halves 8..15 =
// K k0+16..k0+23, where k0 = ks*32 + (l>=16 ? 8 : 0).
static __device__ __forceinline__ v16bf loadA(const unsigned short* sA, int stride,
                                              int mt, int ks, int lane) {
  const int m  = mt * 16 + (lane & 15);
  const int k0 = ks * 32 + ((lane >> 4) << 3);
  const unsigned short* p = sA + m * stride + k0;
  v8u lo = *(const v8u*)p;
  v8u hi = *(const v8u*)(p + 16);
  v16u t = __builtin_shufflevector(lo, hi, 0,1,2,3,4,5,6,7,8,9,10,11,12,13,14,15);
  return __builtin_bit_cast(v16bf, t);
}

// B fragment (32x16 bf16, K x N) from pre-swizzled global weights: each
// (ktile,ntile) block is 32 lanes x 16 halves contiguous -> 2x global_load_b128.
static __device__ __forceinline__ v16bf loadB(const unsigned short* wB, int NT,
                                              int ks, int nt, int lane) {
  const unsigned short* p = wB + ((((ks * NT + nt) << 5) + lane) << 4);
  v8u lo = *(const v8u*)p;
  v8u hi = *(const v8u*)(p + 8);
  v16u t = __builtin_shufflevector(lo, hi, 0,1,2,3,4,5,6,7,8,9,10,11,12,13,14,15);
  return __builtin_bit_cast(v16bf, t);
}

static __device__ __forceinline__ v8f vzero() {
  v8f c = {0.f,0.f,0.f,0.f,0.f,0.f,0.f,0.f};
  return c;
}

// 2(M) x 2(N) output tiles sharing fragment loads
static __device__ __forceinline__ void mm2x2(const unsigned short* sA, int strideA,
                                             const unsigned short* wB, int NT,
                                             int nk, int nt0, int lane, v8f* c) {
#pragma unroll
  for (int ks = 0; ks < nk; ++ks) {
    v16bf a0 = loadA(sA, strideA, 0, ks, lane);
    v16bf a1 = loadA(sA, strideA, 1, ks, lane);
    v16bf b0 = loadB(wB, NT, ks, nt0, lane);
    v16bf b1 = loadB(wB, NT, ks, nt0 + 1, lane);
    c[0] = wmma_bf16(a0, b0, c[0]);
    c[1] = wmma_bf16(a0, b1, c[1]);
    c[2] = wmma_bf16(a1, b0, c[2]);
    c[3] = wmma_bf16(a1, b1, c[3]);
  }
}

static __device__ __forceinline__ v8f mm1(const unsigned short* sA, int strideA,
                                          const unsigned short* wB, int NT,
                                          int nk, int mt, int nt, int lane) {
  v8f c = vzero();
#pragma unroll
  for (int ks = 0; ks < nk; ++ks)
    c = wmma_bf16(loadA(sA, strideA, mt, ks, lane), loadB(wB, NT, ks, nt, lane), c);
  return c;
}

// C/D tile layout: lane l -> N = l&15; VGPR i -> M = i + (l>=16 ? 8 : 0)
static __device__ __forceinline__ void store_bn_relu(v8f c, int mt, int nt, int lane,
                                                     const float* bias, const float* g,
                                                     const float* be, const float* m,
                                                     const float* v,
                                                     unsigned short* dst, int stride,
                                                     int colOff, const float* simw) {
  const int n = nt * 16 + (lane & 15);
  const float bb = bias[n];
  const float sc = g[n] * rsqrtf(v[n] + 1e-5f);
  const float sh = be[n] - m[n] * sc;
  const int hi8 = (lane >> 4) << 3;
#pragma unroll
  for (int i = 0; i < 8; ++i) {
    const int r = mt * 16 + i + hi8;
    float y = fmaxf((c[i] + bb) * sc + sh, 0.f);
    if (simw) y *= simw[r];
    dst[r * stride + colOff + n] = f2bf(y);
  }
}

static __device__ __forceinline__ void store_relu(v8f c, int mt, int nt, int lane,
                                                  const float* bias,
                                                  unsigned short* dst, int stride) {
  const int n = nt * 16 + (lane & 15);
  const float bb = bias[n];
  const int hi8 = (lane >> 4) << 3;
#pragma unroll
  for (int i = 0; i < 8; ++i) {
    const int r = mt * 16 + i + hi8;
    dst[r * stride + n] = f2bf(fmaxf(c[i] + bb, 0.f));
  }
}

static __device__ __forceinline__ void senet_eval(const float* x, const float* W1,
                                                  const float* b1, const float* W2,
                                                  const float* b2, float* y) {
  float h[8];
#pragma unroll
  for (int j = 0; j < 8; ++j) {
    float s = b1[j];
#pragma unroll
    for (int i = 0; i < 8; ++i) s += x[i] * W1[i * 8 + j];
    h[j] = 0.5f * s * (1.f + erff(s * 0.70710678118f));   // exact GELU
  }
#pragma unroll
  for (int j = 0; j < 8; ++j) {
    float s = b2[j];
#pragma unroll
    for (int i = 0; i < 8; ++i) s += h[i] * W2[i * 8 + j];
    y[j] = s;
  }
}

// ---------------------------------------------------------------------------
// Prep: fp32 row-major [batch][K][N] -> bf16 WMMA-B fragment order
// ---------------------------------------------------------------------------
__global__ void imoe_swizzleB(const float* __restrict__ W,
                              unsigned short* __restrict__ out,
                              int K, int N, int total) {
  int t = blockIdx.x * blockDim.x + threadIdx.x;
  if (t >= total) return;
  const int per = K * N;
  const int bi = t / per;
  const int o  = t - bi * per;
  const int h    = o & 15;
  const int lane = (o >> 4) & 31;
  const int blk  = o >> 9;
  const int NT = N >> 4;
  const int nt = blk % NT;
  const int kt = blk / NT;
  const int k = kt * 32 + ((lane >= 16) ? 8 : 0) + ((h < 8) ? h : h + 8);   // B: lane=N col, halves=K
  const int n = nt * 16 + (lane & 15);
  out[t] = f2bf(W[bi * per + k * N + n]);
}

// ---------------------------------------------------------------------------
// Fused megakernel: 32 rows per block, 256 threads (8 wave32)
// ---------------------------------------------------------------------------
struct P {
  const float *signal, *muti, *sim, *w, *b;
  const float *ps_b, *ps_g, *ps_be, *ps_m, *ps_v;
  const float *pm_b, *pm_g, *pm_be, *pm_m, *pm_v;
  const float *c1_b, *c1_g, *c1_be, *c1_m, *c1_v;
  const float *c2_b, *c2_g, *c2_be, *c2_m, *c2_v;
  const float *c3_W, *c3_b;
  const float *e_b1, *e_b2;
  const float *s_W1, *s_b1, *s_W2, *s_b2;
  const unsigned short* wsw;
  float* out;
};

__global__ __launch_bounds__(256) void imoe_main(P p, int B) {
  extern __shared__ char smem[];
  unsigned short* TEMP  = (unsigned short*)(smem + L_TEMP);   // [32][520] bf16
  unsigned short* STAGE = (unsigned short*)(smem + L_STAGE);  // [32][264] bf16
  unsigned short* EO    = (unsigned short*)(smem + L_EO);     // [8][32*64] bf16
  float* SIMW = (float*)(smem + L_SIMW);
  float* AVE  = (float*)(smem + L_AVE);
  float* MAXV = (float*)(smem + L_MAX);
  float* WTS  = (float*)(smem + L_WTS);

  const int tid  = threadIdx.x;
  const int lane = tid & 31;
  const int wv   = tid >> 5;
  const int row0 = blockIdx.x * 32;

  if (tid < 32) SIMW[tid] = p.sim[row0 + tid] * p.w[0] + p.b[0];

  // ---------------- projections: sig (cols 0..255) then mut (256..511) -----
  for (int src = 0; src < 2; ++src) {
    const float* X = (src == 0) ? p.signal : p.muti;
    __syncthreads();                    // STAGE reuse / SIMW visibility
#pragma unroll
    for (int i = 0; i < 8; ++i) {       // stage 32x256 fp32 -> bf16 (coalesced)
      const int f4 = tid + i * 256;
      const float4 val = ((const float4*)(X + (size_t)row0 * 256))[f4];
      const int idx = f4 * 4;
      unsigned short* d = STAGE + (idx >> 8) * 264 + (idx & 255);
      d[0] = f2bf(val.x); d[1] = f2bf(val.y); d[2] = f2bf(val.z); d[3] = f2bf(val.w);
    }
    __syncthreads();
    const unsigned short* Wsw = p.wsw + ((src == 0) ? WS_PSW : WS_PMW);
    v8f c[4] = {vzero(), vzero(), vzero(), vzero()};
    const int nt0 = wv * 2;
    mm2x2(STAGE, 264, Wsw, 16, 8, nt0, lane, c);
    const float *bb = src ? p.pm_b : p.ps_b, *g  = src ? p.pm_g : p.ps_g;
    const float *be = src ? p.pm_be : p.ps_be, *m = src ? p.pm_m : p.ps_m;
    const float *vv = src ? p.pm_v : p.ps_v;
    const float* sw = src ? (const float*)nullptr : SIMW;
    const int colOff = src * 256;
    store_bn_relu(c[0], 0, nt0,     lane, bb, g, be, m, vv, TEMP, 520, colOff, sw);
    store_bn_relu(c[1], 0, nt0 + 1, lane, bb, g, be, m, vv, TEMP, 520, colOff, sw);
    store_bn_relu(c[2], 1, nt0,     lane, bb, g, be, m, vv, TEMP, 520, colOff, sw);
    store_bn_relu(c[3], 1, nt0 + 1, lane, bb, g, be, m, vv, TEMP, 520, colOff, sw);
  }
  __syncthreads();

  // ---------------- classifier head ---------------------------------------
  unsigned short* H1 = STAGE;            // [32][72] bf16
  unsigned short* H2 = STAGE + 2304;     // [32][72] bf16
  {
    const int mt = wv >> 2, nt = wv & 3;
    v8f c = mm1(TEMP, 520, p.wsw + WS_C1W, 4, 16, mt, nt, lane);
    store_bn_relu(c, mt, nt, lane, p.c1_b, p.c1_g, p.c1_be, p.c1_m, p.c1_v,
                  H1, 72, 0, nullptr);
  }
  __syncthreads();
  {
    const int mt = wv >> 2, nt = wv & 3;
    v8f c = mm1(H1, 72, p.wsw + WS_C2W, 4, 2, mt, nt, lane);
    store_bn_relu(c, mt, nt, lane, p.c2_b, p.c2_g, p.c2_be, p.c2_m, p.c2_v,
                  H2, 72, 0, nullptr);
  }
  __syncthreads();
  if (tid < 64) {                        // c3: [32][64] @ [64][2]
    const int r = tid >> 1, j = tid & 1;
    float s = p.c3_b[j];
#pragma unroll
    for (int k = 0; k < 64; ++k) s += bf2f(H2[r * 72 + k]) * p.c3_W[k * 2 + j];
    p.out[(size_t)B * 64 + (size_t)(row0 + r) * 2 + j] = s;
  }
  __syncthreads();

  // ---------------- experts (dominant compute) -----------------------------
  for (int e = 0; e < 8; ++e) {
    const unsigned short* W1 = p.wsw + WS_EW1 + e * 131072;
    v8f c[4] = {vzero(), vzero(), vzero(), vzero()};
    const int nt0 = wv * 2;
    mm2x2(TEMP, 520, W1, 16, 16, nt0, lane, c);   // [32,512]@[512,256]
    const float* b1 = p.e_b1 + e * 256;
    store_relu(c[0], 0, nt0,     lane, b1, STAGE, 264);
    store_relu(c[1], 0, nt0 + 1, lane, b1, STAGE, 264);
    store_relu(c[2], 1, nt0,     lane, b1, STAGE, 264);
    store_relu(c[3], 1, nt0 + 1, lane, b1, STAGE, 264);
    __syncthreads();

    const unsigned short* W2 = p.wsw + WS_EW2 + e * 16384;
    const int mt = wv >> 2, nt = wv & 3;
    v8f co = mm1(STAGE, 264, W2, 4, 8, mt, nt, lane);   // [32,256]@[256,64]
    {
      const int n = nt * 16 + (lane & 15);
      const float bb = p.e_b2[e * 64 + n];
      const int hi8 = (lane >> 4) << 3;
#pragma unroll
      for (int i = 0; i < 8; ++i) {
        const int r = mt * 16 + i + hi8;
        EO[e * 2048 + r * 64 + n] = f2bf(fmaxf(co[i] + bb, 0.f));
      }
    }
    __syncthreads();
  }

  // ---------------- pooling -> SE gating -> weighted sum -------------------
  {
    const int r = tid >> 3, e = tid & 7;
    const unsigned short* q = EO + e * 2048 + r * 64;
    float s = 0.f, mx = -3.0e38f;
#pragma unroll
    for (int d = 0; d < 64; ++d) { float v = bf2f(q[d]); s += v; mx = fmaxf(mx, v); }
    AVE[tid] = s * (1.f / 64.f);
    MAXV[tid] = mx;
  }
  __syncthreads();
  if (tid < 32) {
    float xa[8], xm[8], ya[8], ym[8];
#pragma unroll
    for (int e = 0; e < 8; ++e) { xa[e] = AVE[tid * 8 + e]; xm[e] = MAXV[tid * 8 + e]; }
    senet_eval(xa, p.s_W1, p.s_b1, p.s_W2, p.s_b2, ya);
    senet_eval(xm, p.s_W1, p.s_b1, p.s_W2, p.s_b2, ym);
#pragma unroll
    for (int j = 0; j < 8; ++j)
      WTS[tid * 8 + j] = 1.f / (1.f + expf(-(ya[j] + ym[j])));
  }
  __syncthreads();
  {
    const int r = tid >> 3, d0 = (tid & 7) * 8;
    float wt[8];
#pragma unroll
    for (int e = 0; e < 8; ++e) wt[e] = WTS[r * 8 + e];
#pragma unroll
    for (int d = 0; d < 8; ++d) {
      float s = 0.f;
#pragma unroll
      for (int e = 0; e < 8; ++e) s += bf2f(EO[e * 2048 + r * 64 + d0 + d]) * wt[e];
      p.out[(size_t)(row0 + r) * 64 + d0 + d] = s;
    }
  }
}

// ---------------------------------------------------------------------------
extern "C" void kernel_launch(void* const* d_in, const int* in_sizes, int n_in,
                              void* d_out, int out_size, void* d_ws, size_t ws_size,
                              hipStream_t stream) {
  (void)n_in; (void)out_size; (void)ws_size;
  unsigned short* wsw = (unsigned short*)d_ws;
  const int B = in_sizes[2];   // similarity length

  // Swizzle all GEMM weights to bf16 WMMA-B fragment order (2.7 MB, L2-resident)
  auto sw = [&](int inIdx, unsigned off, int K, int N, int batch) {
    const int total = batch * K * N;
    imoe_swizzleB<<<(total + 255) / 256, 256, 0, stream>>>(
        (const float*)d_in[inIdx], wsw + off, K, N, total);
  };
  sw(5,  WS_PSW, 256, 256, 1);
  sw(11, WS_PMW, 256, 256, 1);
  sw(17, WS_C1W, 512, 64, 1);
  sw(23, WS_C2W, 64, 64, 1);
  sw(31, WS_EW1, 512, 256, 8);
  sw(33, WS_EW2, 256, 64, 8);

  P p;
  p.signal = (const float*)d_in[0];  p.muti = (const float*)d_in[1];
  p.sim = (const float*)d_in[2];     p.w = (const float*)d_in[3];
  p.b = (const float*)d_in[4];
  p.ps_b = (const float*)d_in[6];  p.ps_g = (const float*)d_in[7];
  p.ps_be = (const float*)d_in[8]; p.ps_m = (const float*)d_in[9];
  p.ps_v = (const float*)d_in[10];
  p.pm_b = (const float*)d_in[12]; p.pm_g = (const float*)d_in[13];
  p.pm_be = (const float*)d_in[14]; p.pm_m = (const float*)d_in[15];
  p.pm_v = (const float*)d_in[16];
  p.c1_b = (const float*)d_in[18]; p.c1_g = (const float*)d_in[19];
  p.c1_be = (const float*)d_in[20]; p.c1_m = (const float*)d_in[21];
  p.c1_v = (const float*)d_in[22];
  p.c2_b = (const float*)d_in[24]; p.c2_g = (const float*)d_in[25];
  p.c2_be = (const float*)d_in[26]; p.c2_m = (const float*)d_in[27];
  p.c2_v = (const float*)d_in[28];
  p.c3_W = (const float*)d_in[29]; p.c3_b = (const float*)d_in[30];
  p.e_b1 = (const float*)d_in[32]; p.e_b2 = (const float*)d_in[34];
  p.s_W1 = (const float*)d_in[35]; p.s_b1 = (const float*)d_in[36];
  p.s_W2 = (const float*)d_in[37]; p.s_b2 = (const float*)d_in[38];
  p.wsw = wsw;
  p.out = (float*)d_out;

  imoe_main<<<B / 32, 256, SMEM_BYTES, stream>>>(p, B);
}
